// PolylineNet_14448269984417
// MI455X (gfx1250) — matrix operand order repaired and checked by reading, not compiled, not run on hardware
//
#include <hip/hip_runtime.h>
#include <hip/hip_bf16.h>

// ---------------------------------------------------------------------------
// Fully fused PolylineNet encoder for MI455X (gfx1250, wave32, WMMA).
//   grid: one workgroup per 4 polylines (80 rows = 5 x 16-row WMMA tiles)
//   block: 256 threads = 8 waves; wave w owns output columns [16w, 16w+16)
//   all intermediates in LDS; global traffic = read x once + write out once
//   GEMMs: v_wmma_f32_16x16x32_bf16 (fp32 accumulate, bf16 operands)
//   LayerNorm: wave-parallel, cross-lane shuffle reductions (no LDS serialization)
// ---------------------------------------------------------------------------

typedef __bf16 v16bf __attribute__((ext_vector_type(16)));
typedef __bf16 v8bf  __attribute__((ext_vector_type(8)));
typedef __bf16 v4bf  __attribute__((ext_vector_type(4)));
typedef float  v8f   __attribute__((ext_vector_type(8)));
typedef float  v4f   __attribute__((ext_vector_type(4)));

#define Hh        128
#define PP        20
#define POLYS     4
#define ROWS      (POLYS * PP)      // 80 rows -> exactly 5 row tiles of 16
#define RT        (ROWS / 16)       // 5
#define THREADS   256
#define NWAVES    8
#define SSTRIDE   132               // padded f32 row stride (16B-aligned rows)
#define LNEPS     1e-5f

// LDS carve (bytes, all 16B aligned). Total = 117,344 B (< 320KB/WGP, 2 WG/WGP)
#define OFF_AB    0                         // bf16 activations, up to 80x256
#define OFF_WT    (OFF_AB  + ROWS*256*2)    // 40960 : bf16 transposed weights 128x128
#define OFF_S     (OFF_WT  + 128*128*2)     // 73728 : f32 pre-LN GEMM output 80x132
#define OFF_POOL  (OFF_S   + ROWS*SSTRIDE*4)// 115968: bf16 pooled 4x128
#define OFF_MASK  (OFF_POOL+ POLYS*Hh*2)    // 116992: f32 mask[80] + valid[4]
#define SMEM_BYTES (OFF_MASK + (ROWS + 4)*4 + 16)

// --- A fragment: 16x32 bf16 tile, row-major LDS, stride in elements ---------
// lanes 0-15: row = lane, K 0..7 (v0-3) and 16..23 (v4-7)
// lanes 16-31: row = lane-16, K 8..15 and 24..31
__device__ __forceinline__ v16bf load_a_frag(const __bf16* tileBase, int strideE) {
  const int lane = threadIdx.x & 31;
  const int r    = lane & 15;
  const int koff = (lane < 16) ? 0 : 8;
  const __bf16* p = tileBase + r * strideE + koff;
  v8bf lo = *(const v8bf*)(p);
  v8bf hi = *(const v8bf*)(p + 16);
  v16bf a;
#pragma unroll
  for (int i = 0; i < 8; ++i) { a[i] = lo[i]; a[i + 8] = hi[i]; }
  return a;
}

// --- B fragment: 32x16 bf16 from transposed weights WT[n][k], stride KC -----
// lanes 0-15: col n = lane, K 0..7 and 16..23 ; lanes 16-31: col n = lane-16, K 8..15 and 24..31
__device__ __forceinline__ v16bf load_b_frag(const __bf16* WT, int KC, int kk) {
  const int lane = threadIdx.x & 31;
  const int wave = threadIdx.x >> 5;
  const int n    = (wave << 4) + (lane & 15);
  const int koff = (lane < 16) ? 0 : 8;
  const __bf16* p = WT + n * KC + kk + koff;
  v8bf lo = *(const v8bf*)(p);
  v8bf hi = *(const v8bf*)(p + 16);
  v16bf b;
#pragma unroll
  for (int i = 0; i < 8; ++i) { b[i] = lo[i]; b[i + 8] = hi[i]; }
  return b;
}

template <int NRT>
__device__ __forceinline__ void gemm_acc(const __bf16* A, int aStride,
                                         const __bf16* WT, int KC, v8f acc[NRT]) {
  for (int kk = 0; kk < KC; kk += 32) {
    v16bf b = load_b_frag(WT, KC, kk);
#pragma unroll
    for (int rt = 0; rt < NRT; ++rt) {
      v16bf a = load_a_frag(A + (rt * 16) * aStride + kk, aStride);
      acc[rt] = __builtin_amdgcn_wmma_f32_16x16x32_bf16(
          /*neg_a=*/false, a, /*neg_b=*/false, b,
          /*c_mod=*/(short)0, acc[rt], /*reuse_a=*/false, /*reuse_b=*/false);
    }
  }
}

// C/D layout: element e of v8f -> row = rt*16 + e + (lane<16 ? 0 : 8); col = 16*wave + lane%16
template <int NRT>
__device__ __forceinline__ void store_acc_bias(float* S, const v8f acc[NRT],
                                               const float* bias, int bOff) {
  const int lane = threadIdx.x & 31;
  const int wave = threadIdx.x >> 5;
  const int col  = (wave << 4) + (lane & 15);
  const float b  = bias[bOff + col];
  const int rb   = (lane < 16) ? 0 : 8;
#pragma unroll
  for (int rt = 0; rt < NRT; ++rt)
#pragma unroll
    for (int e = 0; e < 8; ++e)
      S[(rt * 16 + rb + e) * SSTRIDE + col] = acc[rt][e] + b;
}

// Stage W[kOff+k][nOff+n] (global row-major KxNw f32) transposed -> WT[n][k] bf16,
// n in [0,128), k in [0,KC), zero-padding rows past K (fc1_W1: K=13 -> KC=32).
__device__ __forceinline__ void stage_wt(__bf16* WT, const float* __restrict__ W,
                                         int K, int Nw, int kOff, int KC, int nOff) {
  for (int i = threadIdx.x; i < 128 * KC; i += THREADS) {
    const int n = i / KC, k = i - n * KC;
    const int kg = kOff + k;
    const float v = (kg < K) ? W[(size_t)kg * Nw + nOff + n] : 0.0f;
    WT[i] = (__bf16)v;
  }
}

// Wave-parallel LayerNorm + ReLU (+ optional point mask) -> bf16 dst.
// Wave handles rows {wave, wave+8, ...}; lane owns 4 of the 128 columns.
// Stats via 5-step __shfl_xor cross-lane reduction (single-pass E[x^2] form).
__device__ __forceinline__ void ln_relu_wave(const float* S, __bf16* dst, int dstStride,
                                             const float* __restrict__ g,
                                             const float* __restrict__ be,
                                             const float* maskv, int rows) {
  const int wave = threadIdx.x >> 5;
  const int lane = threadIdx.x & 31;
  const int k0   = lane << 2;                 // 4 columns per lane
  const v4f gv = *(const v4f*)(g + k0);
  const v4f bv = *(const v4f*)(be + k0);
  for (int r = wave; r < rows; r += NWAVES) {
    const v4f x = *(const v4f*)(S + r * SSTRIDE + k0);   // one ds_load_b128
    float s = x[0] + x[1] + x[2] + x[3];
    float q = x[0]*x[0] + x[1]*x[1] + x[2]*x[2] + x[3]*x[3];
#pragma unroll
    for (int m = 16; m >= 1; m >>= 1) {
      s += __shfl_xor(s, m, 32);
      q += __shfl_xor(q, m, 32);
    }
    const float mu  = s * (1.0f / Hh);
    const float var = fmaxf(q * (1.0f / Hh) - mu * mu, 0.0f);
    const float inv = rsqrtf(var + LNEPS);
    const float mk  = maskv ? maskv[r] : 1.0f;
    v4bf o;
#pragma unroll
    for (int j = 0; j < 4; ++j) {
      const float y = (x[j] - mu) * inv * gv[j] + bv[j];
      o[j] = (__bf16)(fmaxf(y, 0.0f) * mk);
    }
    *(v4bf*)(dst + r * dstStride + k0) = o;              // one packed 8B store
  }
}

__global__ __launch_bounds__(THREADS) void polyline_fused_kernel(
    const float* __restrict__ X, const unsigned char* __restrict__ Mk,
    const float* fc1W1, const float* fc1b1, const float* fc1g1, const float* fc1be1,
    const float* fc1W2, const float* fc1b2, const float* fc1g2, const float* fc1be2,
    const float* fc2W1, const float* fc2b1, const float* fc2g1, const float* fc2be1,
    const float* fc2W2, const float* fc2b2, const float* fc2g2, const float* fc2be2,
    const float* foW1, const float* fob1, const float* foW2, const float* fob2,
    float* __restrict__ Out) {
  extern __shared__ __align__(16) unsigned char smem[];
  __bf16* Abuf   = (__bf16*)(smem + OFF_AB);    // activations (x / h / cat / foA)
  __bf16* WT     = (__bf16*)(smem + OFF_WT);    // transposed weight chunk 128x128
  float*  S      = (float*) (smem + OFF_S);     // pre-LN GEMM output (fp32)
  __bf16* pooled = (__bf16*)(smem + OFF_POOL);  // 4x128 max-pooled features
  float*  maskv  = (float*) (smem + OFF_MASK);  // per-point mask as 0/1 float
  float*  validv = maskv + ROWS;                // per-polyline any(mask)

  const int tid = threadIdx.x;
  const size_t gg = blockIdx.x;                 // group of 4 polylines

  // ---- load point mask + raw points (80 x 13, K-padded to 32) --------------
  if (tid < ROWS) maskv[tid] = Mk[gg * ROWS + tid] ? 1.0f : 0.0f;
  const float* Xg = X + gg * (size_t)(ROWS * 13);
  for (int i = tid; i < ROWS * 32; i += THREADS) {
    const int r = i >> 5, k = i & 31;
    Abuf[i] = (__bf16)((k < 13) ? Xg[r * 13 + k] : 0.0f);
  }
  __syncthreads();
  if (tid < POLYS) {
    float v = 0.f;
    for (int r = 0; r < PP; ++r) v = fmaxf(v, maskv[tid * PP + r]);
    validv[tid] = v;
  }

  const v8f vzero = {0.f, 0.f, 0.f, 0.f, 0.f, 0.f, 0.f, 0.f};
  v8f acc[RT];

  // ==== fc1 layer 1: [80x13] @ [13x128] ====================================
  stage_wt(WT, fc1W1, 13, 128, 0, 32, 0);
  __syncthreads();
#pragma unroll
  for (int t = 0; t < RT; ++t) acc[t] = vzero;
  gemm_acc<RT>(Abuf, 32, WT, 32, acc);
  store_acc_bias<RT>(S, acc, fc1b1, 0);
  __syncthreads();
  ln_relu_wave(S, Abuf, Hh, fc1g1, fc1be1, nullptr, ROWS);
  __syncthreads();

  // ==== fc1 layer 2: [80x128] @ [128x128]; masked feat -> cat cols [0,128) ==
  stage_wt(WT, fc1W2, 128, 128, 0, 128, 0);
  __syncthreads();
#pragma unroll
  for (int t = 0; t < RT; ++t) acc[t] = vzero;
  gemm_acc<RT>(Abuf, 128, WT, 128, acc);
  store_acc_bias<RT>(S, acc, fc1b2, 0);
  __syncthreads();
  ln_relu_wave(S, Abuf, 256, fc1g2, fc1be2, maskv, ROWS);
  __syncthreads();

  // ==== max-pool over the 20 points, broadcast into cat cols [128,256) =====
  for (int i = tid; i < POLYS * Hh; i += THREADS) {
    const int p = i >> 7, c = i & 127;
    float m = 0.f;  // ReLU+mask => values >= 0, matches reference max
    for (int r = 0; r < PP; ++r) m = fmaxf(m, (float)Abuf[(p * PP + r) * 256 + c]);
    pooled[i] = (__bf16)m;
  }
  __syncthreads();
  for (int i = tid; i < ROWS * (Hh / 4); i += THREADS) {  // 8B packed copies
    const int r = i >> 5, c4 = (i & 31) << 2;
    *(v4bf*)(Abuf + r * 256 + 128 + c4) = *(const v4bf*)(pooled + (r / PP) * Hh + c4);
  }
  __syncthreads();

  // ==== fc2 layer 1: [80x256] @ [256x128], K split in two 128 stages ========
#pragma unroll
  for (int t = 0; t < RT; ++t) acc[t] = vzero;
  for (int ks = 0; ks < 2; ++ks) {
    stage_wt(WT, fc2W1, 256, 128, ks * 128, 128, 0);
    __syncthreads();
    gemm_acc<RT>(Abuf + ks * 128, 256, WT, 128, acc);
    __syncthreads();  // all waves done reading WT before restage
  }
  store_acc_bias<RT>(S, acc, fc2b1, 0);
  __syncthreads();
  ln_relu_wave(S, Abuf, Hh, fc2g1, fc2be1, nullptr, ROWS);
  __syncthreads();

  // ==== fc2 layer 2: [80x128] @ [128x128]; masked feat2 =====================
  stage_wt(WT, fc2W2, 128, 128, 0, 128, 0);
  __syncthreads();
#pragma unroll
  for (int t = 0; t < RT; ++t) acc[t] = vzero;
  gemm_acc<RT>(Abuf, 128, WT, 128, acc);
  store_acc_bias<RT>(S, acc, fc2b2, 0);
  __syncthreads();
  ln_relu_wave(S, Abuf, Hh, fc2g2, fc2be2, maskv, ROWS);
  __syncthreads();

  // ==== max-pool 2 -> pooled2 [4x128] =======================================
  for (int i = tid; i < POLYS * Hh; i += THREADS) {
    const int p = i >> 7, c = i & 127;
    float m = 0.f;
    for (int r = 0; r < PP; ++r) m = fmaxf(m, (float)Abuf[(p * PP + r) * Hh + c]);
    pooled[i] = (__bf16)m;
  }
  __syncthreads();

  // ==== head A-tile: 16x128, rows 0..3 = pooled2, rest zero =================
  for (int i = tid; i < 16 * Hh; i += THREADS) {
    const int r = i >> 7;
    Abuf[i] = (r < POLYS) ? pooled[(r << 7) + (i & 127)] : (__bf16)0.0f;
  }
  __syncthreads();

  // ==== fo layer 1: relu([16x128] @ [128x128] + b) ==========================
  stage_wt(WT, foW1, 128, 128, 0, 128, 0);
  __syncthreads();
  {
    v8f a1[1] = {vzero};
    gemm_acc<1>(Abuf, 128, WT, 128, a1);
    store_acc_bias<1>(S, a1, fob1, 0);
  }
  __syncthreads();
  for (int i = tid; i < 16 * Hh; i += THREADS) {
    const int r = i >> 7, k = i & 127;
    Abuf[i] = (__bf16)fmaxf(S[r * SSTRIDE + k], 0.0f);
  }
  __syncthreads();

  // ==== fo layer 2: [16x128] @ [128x256] in two N halves; store masked ======
  const int lane = tid & 31, wave = tid >> 5;
  for (int np = 0; np < 2; ++np) {
    stage_wt(WT, foW2, 128, 256, 0, 128, np * 128);
    __syncthreads();
    v8f a2[1] = {vzero};
    gemm_acc<1>(Abuf, 128, WT, 128, a2);
    if (lane < 16) {  // lanes 0-15 hold rows 0..7; only rows 0..3 are real
      const int col = np * 128 + (wave << 4) + lane;
      const float b = fob2[col];
#pragma unroll
      for (int e = 0; e < POLYS; ++e) {
        Out[(gg * POLYS + e) * 256 + col] = (a2[0][e] + b) * validv[e];
      }
    }
    __syncthreads();  // guard WT restage vs. in-flight gemm reads
  }
}

extern "C" void kernel_launch(void* const* d_in, const int* in_sizes, int n_in,
                              void* d_out, int out_size, void* d_ws, size_t ws_size,
                              hipStream_t stream) {
  (void)in_sizes; (void)n_in; (void)out_size; (void)d_ws; (void)ws_size;
  const float*         X  = (const float*)d_in[0];
  const unsigned char* Mk = (const unsigned char*)d_in[1];  // numpy bool = 1 byte
  const float* fc1W1 = (const float*)d_in[2],  *fc1b1 = (const float*)d_in[3];
  const float* fc1g1 = (const float*)d_in[4],  *fc1be1= (const float*)d_in[5];
  const float* fc1W2 = (const float*)d_in[6],  *fc1b2 = (const float*)d_in[7];
  const float* fc1g2 = (const float*)d_in[8],  *fc1be2= (const float*)d_in[9];
  const float* fc2W1 = (const float*)d_in[10], *fc2b1 = (const float*)d_in[11];
  const float* fc2g1 = (const float*)d_in[12], *fc2be1= (const float*)d_in[13];
  const float* fc2W2 = (const float*)d_in[14], *fc2b2 = (const float*)d_in[15];
  const float* fc2g2 = (const float*)d_in[16], *fc2be2= (const float*)d_in[17];
  const float* foW1  = (const float*)d_in[18], *fob1  = (const float*)d_in[19];
  const float* foW2  = (const float*)d_in[20], *fob2  = (const float*)d_in[21];
  float* Out = (float*)d_out;

  static_assert(SMEM_BYTES < 320 * 1024, "fits one WGP LDS");
  hipFuncSetAttribute((const void*)polyline_fused_kernel,
                      hipFuncAttributeMaxDynamicSharedMemorySize, SMEM_BYTES);

  const int nGroups = (32 * 512) / POLYS;  // 4096 workgroups of 4 polylines
  polyline_fused_kernel<<<nGroups, THREADS, SMEM_BYTES, stream>>>(
      X, Mk,
      fc1W1, fc1b1, fc1g1, fc1be1, fc1W2, fc1b2, fc1g2, fc1be2,
      fc2W1, fc2b1, fc2g1, fc2be1, fc2W2, fc2b2, fc2g2, fc2be2,
      foW1, fob1, foW2, fob2, Out);
}
